// Adaptive_pooling_33346126086887
// MI455X (gfx1250) — matrix operand order repaired and checked by reading, not compiled, not run on hardware
//
#include <hip/hip_runtime.h>

// ---------------------------------------------------------------------------
// CDNA5 (gfx1250) implementation of the adaptive-pooling reference.
// All O(N^3)/O(N^2 D) matmuls run through v_wmma_f32_16x16x32_bf16.
// ---------------------------------------------------------------------------

typedef __attribute__((ext_vector_type(16))) __bf16 v16bf;
typedef __attribute__((ext_vector_type(8)))  __bf16 v8bf;
typedef __attribute__((ext_vector_type(8)))  float  v8f;

static constexpr int    Nn = 4096;
static constexpr int    Dd = 256;
static constexpr int    Ee = 65536;
static constexpr size_t NN = (size_t)Nn * (size_t)Nn;

// ---- bf16 helpers ---------------------------------------------------------
__device__ __forceinline__ __bf16 f2bf(float f) {
  unsigned u = __float_as_uint(f);
  unsigned r = u + 0x7FFFu + ((u >> 16) & 1u);   // round-to-nearest-even
  unsigned short s = (unsigned short)(r >> 16);
  return __builtin_bit_cast(__bf16, s);
}
__device__ __forceinline__ float bf2f(__bf16 b) {
  unsigned short s = __builtin_bit_cast(unsigned short, b);
  return __uint_as_float(((unsigned)s) << 16);
}

__device__ __forceinline__ void atomicMaxF(float* addr, float v) {
  unsigned* ua = (unsigned*)addr;
  unsigned old = *ua;
  while (__uint_as_float(old) < v) {
    unsigned assumed = old;
    old = atomicCAS(ua, assumed, __float_as_uint(v));
    if (old == assumed) break;
  }
}

// ---- block reductions (blockDim.x == 256) ---------------------------------
__device__ __forceinline__ float blk_sum(float v, float* sd) {
  int t = threadIdx.x;
  sd[t] = v; __syncthreads();
  for (int s = 128; s > 0; s >>= 1) { if (t < s) sd[t] += sd[t + s]; __syncthreads(); }
  float r = sd[0]; __syncthreads();
  return r;
}
__device__ __forceinline__ float blk_max(float v, float* sd) {
  int t = threadIdx.x;
  sd[t] = v; __syncthreads();
  for (int s = 128; s > 0; s >>= 1) { if (t < s) sd[t] = fmaxf(sd[t], sd[t + s]); __syncthreads(); }
  float r = sd[0]; __syncthreads();
  return r;
}

// ---- WMMA fragment loads (layouts per CDNA5 ISA 7.12.2, wave32) -----------
// A (16x32 bf16, MxK): lane L -> row m=L&15; h=L>>4.
//   elems[0..7]  = A[m, k0+8h .. k0+8h+7]
//   elems[8..15] = A[m, k0+16+8h .. ]
__device__ __forceinline__ v16bf loadA(const __bf16* row_k0, int h) {
  v8bf lo = *reinterpret_cast<const v8bf*>(row_k0 + 8 * h);
  v8bf hi = *reinterpret_cast<const v8bf*>(row_k0 + 16 + 8 * h);
  return __builtin_shufflevector(lo, hi, 0,1,2,3,4,5,6,7,8,9,10,11,12,13,14,15);
}
// B (32x16 bf16, KxN) loaded from Bt (row-major NxK): lane L -> col n=L&15.
//   elems[0..7] = Bt[n, k0+8h ..], elems[8..15] = Bt[n, k0+16+8h ..]  (same k order as A)
__device__ __forceinline__ v16bf loadB(const __bf16* row_k0, int h) {
  v8bf lo = *reinterpret_cast<const v8bf*>(row_k0 + 8 * h);
  v8bf hi = *reinterpret_cast<const v8bf*>(row_k0 + 16 + 8 * h);
  return __builtin_shufflevector(lo, hi, 0,1,2,3,4,5,6,7,8,9,10,11,12,13,14,15);
}

__device__ __forceinline__ void store_tile(float* C, float* Cadj, v8f acc,
                                           int tm, int tn, int ml, int h, int ldc) {
#pragma unroll
  for (int r = 0; r < 8; ++r) {           // C/D layout: VGPR r -> M = 8*h + r, N = lane&15
    int mm = tm + 8 * h + r;
    size_t off = (size_t)mm * ldc + tn + ml;
    float v = acc[r];
    C[off] = v;
    if (Cadj) Cadj[off] = (v > 0.f) ? 1.f : 0.f;
  }
}

// ---- generic C[MxNc] = A[MxK] * Bt[NcxK]^T, bf16 in / f32 out -------------
// block = 256 threads = 8 waves in 2x4 grid; wave tile = 32x32 (2x2 WMMA).
// grid = (Nc/128, M/64)
__global__ __launch_bounds__(256)
void gemm_bf16_ABt(const __bf16* __restrict__ A, const __bf16* __restrict__ Bt,
                   float* __restrict__ C, float* __restrict__ Cadj,
                   int M, int Nc, int K, int lda, int ldb, int ldc) {
  const int lane = threadIdx.x & 31;
  const int wave = threadIdx.x >> 5;
  const int wr = wave >> 2;              // 0..1
  const int wc = wave & 3;               // 0..3
  const int m0 = blockIdx.y * 64 + wr * 32;
  const int n0 = blockIdx.x * 128 + wc * 32;
  const int ml = lane & 15;
  const int h  = lane >> 4;

  const __bf16* a0p = A  + (size_t)(m0 + ml)      * lda;
  const __bf16* a1p = A  + (size_t)(m0 + 16 + ml) * lda;
  const __bf16* b0p = Bt + (size_t)(n0 + ml)      * ldb;
  const __bf16* b1p = Bt + (size_t)(n0 + 16 + ml) * ldb;

  v8f acc00{}, acc01{}, acc10{}, acc11{};
  for (int k0 = 0; k0 < K; k0 += 32) {
    __builtin_prefetch(a0p + k0 + 128, 0, 1);
    __builtin_prefetch(b0p + k0 + 128, 0, 1);
    v16bf a0 = loadA(a0p + k0, h);
    v16bf a1 = loadA(a1p + k0, h);
    v16bf b0 = loadB(b0p + k0, h);
    v16bf b1 = loadB(b1p + k0, h);
    acc00 = __builtin_amdgcn_wmma_f32_16x16x32_bf16(false, a0, false, b0, (short)0, acc00, false, false);
    acc01 = __builtin_amdgcn_wmma_f32_16x16x32_bf16(false, a0, false, b1, (short)0, acc01, false, false);
    acc10 = __builtin_amdgcn_wmma_f32_16x16x32_bf16(false, a1, false, b0, (short)0, acc10, false, false);
    acc11 = __builtin_amdgcn_wmma_f32_16x16x32_bf16(false, a1, false, b1, (short)0, acc11, false, false);
  }
  store_tile(C, Cadj, acc00, m0,      n0,      ml, h, ldc);
  store_tile(C, Cadj, acc01, m0,      n0 + 16, ml, h, ldc);
  store_tile(C, Cadj, acc10, m0 + 16, n0,      ml, h, ldc);
  store_tile(C, Cadj, acc11, m0 + 16, n0 + 16, ml, h, ldc);
}

// ---- per-node precompute: normalize, a/b projections, emb^T, init ---------
__global__ __launch_bounds__(256)
void k_pre(const float* __restrict__ emb, const float* __restrict__ W,
           float* a_vec, float* b_vec, float* mseg, float* den,
           __bf16* Zbf, __bf16* embT) {
  __shared__ float sd[256];
  const int i = blockIdx.x, t = threadIdx.x;
  float x = emb[(size_t)i * Dd + t];
  float ss = blk_sum(x * x, sd);
  float nrm = fmaxf(sqrtf(ss), 1e-12f);
  Zbf[(size_t)i * Dd + t] = f2bf(x / nrm);
  embT[(size_t)t * Nn + i] = f2bf(x);
  float av = blk_sum(x * W[t], sd);
  float bv = blk_sum(x * W[Dd + t], sd);
  if (t == 0) { a_vec[i] = av; b_vec[i] = bv; mseg[i] = -1e30f; den[i] = 0.f; }
}

// ---- edge attention (segment softmax over src) ----------------------------
__global__ void k_edge1(const int* __restrict__ ei, const float* a_vec, const float* b_vec,
                        const float* bsc, float* s_edge, float* mseg) {
  int e = blockIdx.x * 256 + threadIdx.x;
  if (e >= Ee) return;
  int src = ei[e], dst = ei[Ee + e];
  float s = a_vec[src] + b_vec[dst] + bsc[0];
  s = (s > 0.f) ? s : 0.01f * s;              // leaky_relu
  s_edge[e] = s;
  atomicMaxF(&mseg[src], s);
}
__global__ void k_edge2(const int* __restrict__ ei, float* s_edge,
                        const float* mseg, float* den) {
  int e = blockIdx.x * 256 + threadIdx.x;
  if (e >= Ee) return;
  int src = ei[e];
  float ex = expf(s_edge[e] - mseg[src]);
  s_edge[e] = ex;
  atomicAdd(&den[src], ex);
}
__global__ void k_edge3(const int* __restrict__ ei, const float* s_edge,
                        const float* den, float* fitness) {
  int e = blockIdx.x * 256 + threadIdx.x;
  if (e >= Ee) return;
  int src = ei[e], dst = ei[Ee + e];
  atomicAdd(&fitness[(size_t)src * Nn + dst], s_edge[e] / den[src]);
}

// ---- base = offdiag(edge) & (fitness >= 0.1); also write base^T -----------
__global__ void k_base(const float* __restrict__ fit, const float* __restrict__ edgeM,
                       __bf16* base, __bf16* baseT) {
  __shared__ __bf16 tile[64][66];
  const int j0 = blockIdx.x * 64, i0 = blockIdx.y * 64, tx = threadIdx.x;
  for (int rr = threadIdx.y; rr < 64; rr += 4) {
    int i = i0 + rr, j = j0 + tx;
    size_t idx = (size_t)i * Nn + j;
    float v = (edgeM[idx] != 0.f && i != j && fit[idx] >= 0.1f) ? 1.f : 0.f;
    __bf16 bv = f2bf(v);
    base[idx] = bv;
    tile[rr][tx] = bv;
  }
  __syncthreads();
  for (int rr = threadIdx.y; rr < 64; rr += 4)
    baseT[(size_t)(j0 + rr) * Nn + i0 + tx] = tile[tx][rr];
}

// ---- A2 = relu(offdiag(base@base > 0.5) - base), stored into G in place ---
__global__ void k_A2(float* __restrict__ G, const __bf16* __restrict__ base) {
  size_t idx = (size_t)blockIdx.x * 256 + threadIdx.x;
  int i = (int)(idx >> 12), j = (int)(idx & 4095);
  float a2 = (G[idx] > 0.5f && i != j) ? 1.f : 0.f;
  if (bf2f(base[idx]) != 0.f) a2 = 0.f;
  G[idx] = a2;
}

// ---- per-row cluster scores ----------------------------------------------
__global__ __launch_bounds__(256)
void k_scores(const float* __restrict__ fit, const __bf16* __restrict__ base,
              const float* __restrict__ A2, float* cs) {
  __shared__ float sd[256];
  const int i = blockIdx.x;
  float n1 = 0.f, d1 = 0.f, n2 = 0.f, d2 = 0.f;
  for (int j = threadIdx.x; j < Nn; j += 256) {
    size_t idx = (size_t)i * Nn + j;
    float f = fit[idx], b = bf2f(base[idx]), a2 = A2[idx];
    n1 += f * b;  d1 += b;  n2 += f * a2;  d2 += a2;
  }
  n1 = blk_sum(n1, sd); d1 = blk_sum(d1, sd);
  n2 = blk_sum(n2, sd); d2 = blk_sum(d2, sd);
  if (threadIdx.x == 0) {
    float s1 = (d1 > 0.f) ? n1 / d1 : 0.f;
    float s2 = (d2 > 0.f) ? n2 / d2 : 0.f;
    cs[i] = 0.5f * (s1 + s2);
  }
}

// ---- cluster mask: cs[i] > max_j(base[i,j]*cs[j])  (0 included via diag) --
__global__ __launch_bounds__(256)
void k_mask(const __bf16* __restrict__ base, const float* __restrict__ cs,
            unsigned* mask, float* out_mask) {
  __shared__ float sd[256];
  const int i = blockIdx.x;
  float mx = 0.f;
  for (int j = threadIdx.x; j < Nn; j += 256) {
    if (bf2f(base[(size_t)i * Nn + j]) != 0.f) mx = fmaxf(mx, cs[j]);
  }
  mx = blk_max(mx, sd);
  if (threadIdx.x == 0) {
    bool mk = cs[i] > mx;
    mask[i] = mk ? 1u : 0u;
    out_mask[i] = mk ? 1.f : 0.f;
  }
}

// ---- column sums of S0 = base | A2 ---------------------------------------
__global__ void k_colsum(const __bf16* __restrict__ base, const float* __restrict__ A2,
                         unsigned* colsum) {
  int j = blockIdx.x * 256 + threadIdx.x;
  unsigned c = 0;
  for (int i = 0; i < Nn; ++i) {
    size_t idx = (size_t)i * Nn + j;
    if (bf2f(base[idx]) != 0.f || A2[idx] != 0.f) ++c;
  }
  colsum[j] = c;
}

// ---- reduced[i] = any_j(S0[i,j] & mask[j]) | (colsum[i]==0) ---------------
__global__ __launch_bounds__(256)
void k_reduced(const __bf16* __restrict__ base, const float* __restrict__ A2,
               const unsigned* __restrict__ mask, const unsigned* __restrict__ colsum,
               unsigned* reduced) {
  __shared__ float sd[256];
  const int i = blockIdx.x;
  float any = 0.f;
  for (int j = threadIdx.x; j < Nn; j += 256) {
    size_t idx = (size_t)i * Nn + j;
    bool s0 = (bf2f(base[idx]) != 0.f) || (A2[idx] != 0.f);
    if (s0 && mask[j]) any = 1.f;
  }
  any = blk_max(any, sd);
  if (threadIdx.x == 0) reduced[i] = (any > 0.f || colsum[i] == 0u) ? 1u : 0u;
}

// ---- build S / S_w (B output), write S^T, S_w^T (bf16), EMw (bf16) --------
// NOTE: EMw overwrites the base buffer; each thread reads base[idx] first.
__global__ void k_S(const float* __restrict__ fit, const float* __restrict__ A2,
                    __bf16* __restrict__ base_then_EMw, const float* __restrict__ edgeW,
                    const unsigned* __restrict__ mask, const unsigned* __restrict__ reduced,
                    float* __restrict__ outB, __bf16* ST, __bf16* SwT) {
  __shared__ __bf16 tS[64][66];
  __shared__ __bf16 tSw[64][66];
  const int j0 = blockIdx.x * 64, i0 = blockIdx.y * 64, tx = threadIdx.x;
  const int j = j0 + tx;
  const float keep_col = reduced[j] ? 0.f : 1.f;
  const float not_keep = (mask[j] || reduced[j]) ? 1.f : 0.f;
  for (int rr = threadIdx.y; rr < 64; rr += 4) {
    int i = i0 + rr;
    size_t idx = (size_t)i * Nn + j;
    float bse = bf2f(base_then_EMw[idx]);
    float s0 = (bse != 0.f || A2[idx] != 0.f) ? 1.f : 0.f;
    float Sv  = ((i == j) ? 1.f : s0 * not_keep) * keep_col;
    float Swv = ((i == j) ? 1.f : s0 * fit[idx] * not_keep) * keep_col;
    outB[idx] = Swv;
    base_then_EMw[idx] = f2bf((i == j) ? 1.f : edgeW[idx]);   // EMw
    tS[rr][tx]  = f2bf(Sv);
    tSw[rr][tx] = f2bf(Swv);
  }
  __syncthreads();
  for (int rr = threadIdx.y; rr < 64; rr += 4) {
    size_t o = (size_t)(j0 + rr) * Nn + i0 + tx;
    ST[o]  = tS[tx][rr];
    SwT[o] = tSw[tx][rr];
  }
}

// ---- f32 -> bf16 tiled transpose (U -> U^T) -------------------------------
__global__ void k_transU(const float* __restrict__ U, __bf16* UT) {
  __shared__ __bf16 tile[64][66];
  const int j0 = blockIdx.x * 64, i0 = blockIdx.y * 64, tx = threadIdx.x;
  for (int rr = threadIdx.y; rr < 64; rr += 4)
    tile[rr][tx] = f2bf(U[(size_t)(i0 + rr) * Nn + j0 + tx]);
  __syncthreads();
  for (int rr = threadIdx.y; rr < 64; rr += 4)
    UT[(size_t)(j0 + rr) * Nn + i0 + tx] = tile[tx][rr];
}

// ---------------------------------------------------------------------------
extern "C" void kernel_launch(void* const* d_in, const int* in_sizes, int n_in,
                              void* d_out, int out_size, void* d_ws, size_t ws_size,
                              hipStream_t stream) {
  (void)in_sizes; (void)n_in; (void)out_size; (void)ws_size;
  const float* emb   = (const float*)d_in[0];
  const int*   eidx  = (const int*)d_in[1];
  const float* edgeM = (const float*)d_in[2];
  const float* edgeW = (const float*)d_in[3];
  const float* Wsc   = (const float*)d_in[4];
  const float* bsc   = (const float*)d_in[5];

  // workspace carve
  char* ws = (char*)d_ws;
  size_t o = 0;
  float*  fitness = (float*)(ws + o);  o += NN * 4;               // f32 NxN
  float*  G       = (float*)(ws + o);  o += NN * 4;               // f32 NxN (base@base -> A2 -> U)
  __bf16* bf1     = (__bf16*)(ws + o); o += NN * 2;               // base -> EMw
  __bf16* bf2     = (__bf16*)(ws + o); o += NN * 2;               // base^T -> S^T
  __bf16* bf3     = (__bf16*)(ws + o); o += NN * 2;               // S_w^T -> U^T
  __bf16* Zbf     = (__bf16*)(ws + o); o += (size_t)Nn * Dd * 2;  // normalized emb
  __bf16* embT    = (__bf16*)(ws + o); o += (size_t)Nn * Dd * 2;  // emb^T
  float*  s_edge  = (float*)(ws + o);  o += (size_t)Ee * 4;
  float*  a_vec   = (float*)(ws + o);  o += (size_t)Nn * 4;
  float*  b_vec   = (float*)(ws + o);  o += (size_t)Nn * 4;
  float*  mseg    = (float*)(ws + o);  o += (size_t)Nn * 4;
  float*  den     = (float*)(ws + o);  o += (size_t)Nn * 4;
  float*  cs      = (float*)(ws + o);  o += (size_t)Nn * 4;
  unsigned* mask    = (unsigned*)(ws + o); o += (size_t)Nn * 4;
  unsigned* reduced = (unsigned*)(ws + o); o += (size_t)Nn * 4;
  unsigned* colsum  = (unsigned*)(ws + o); o += (size_t)Nn * 4;

  // output carve: (pooled[N,D], new_adj[N,N], new_w[N,N], B[N,N], mask[N])
  float* out        = (float*)d_out;
  float* out_pooled = out;
  float* out_adj    = out_pooled + (size_t)Nn * Dd;
  float* out_neww   = out_adj + NN;
  float* out_B      = out_neww + NN;
  float* out_mask   = out_B + NN;

  const dim3 blk256(256);
  const dim3 tblk(64, 4);
  const dim3 tgrid(Nn / 64, Nn / 64);
  const dim3 gemmNN(Nn / 128, Nn / 64);

  // 1) per-node precompute
  k_pre<<<Nn, blk256, 0, stream>>>(emb, Wsc, a_vec, b_vec, mseg, den, Zbf, embT);
  // 2) connectivity fitness = Z Z^T  (A x B^T with A=B=Z)
  gemm_bf16_ABt<<<gemmNN, blk256, 0, stream>>>(Zbf, Zbf, fitness, nullptr,
                                               Nn, Nn, Dd, Dd, Dd, Nn);
  // 3) edge attention (segment softmax) scattered into fitness
  k_edge1<<<Ee / 256, blk256, 0, stream>>>(eidx, a_vec, b_vec, bsc, s_edge, mseg);
  k_edge2<<<Ee / 256, blk256, 0, stream>>>(eidx, s_edge, mseg, den);
  k_edge3<<<Ee / 256, blk256, 0, stream>>>(eidx, s_edge, den, fitness);
  // 4) base matrix + transpose
  k_base<<<tgrid, tblk, 0, stream>>>(fitness, edgeM, bf1, bf2);
  // 5) G = base @ base
  gemm_bf16_ABt<<<gemmNN, blk256, 0, stream>>>(bf1, bf2, G, nullptr,
                                               Nn, Nn, Nn, Nn, Nn, Nn);
  // 6) A2 = relu(offdiag(G>0.5) - base)  (in place in G)
  k_A2<<<(unsigned)(NN / 256), blk256, 0, stream>>>(G, bf1);
  // 7) cluster scores, mask, reduced
  k_scores<<<Nn, blk256, 0, stream>>>(fitness, bf1, G, cs);
  k_mask<<<Nn, blk256, 0, stream>>>(bf1, cs, mask, out_mask);
  k_colsum<<<Nn / 256, blk256, 0, stream>>>(bf1, G, colsum);
  k_reduced<<<Nn, blk256, 0, stream>>>(bf1, G, mask, colsum, reduced);
  // 8) S / S_w (B output), S^T, S_w^T, EMw
  k_S<<<tgrid, tblk, 0, stream>>>(fitness, G, bf1, edgeW, mask, reduced,
                                  out_B, bf2, bf3);
  // 9) pooled = S_w^T @ emb  (A=S_w^T, B^T=emb^T)
  gemm_bf16_ABt<<<dim3(Dd / 128, Nn / 64), blk256, 0, stream>>>(bf3, embT, out_pooled, nullptr,
                                                                Nn, Dd, Nn, Nn, Nn, Dd);
  // 10) U = EMw @ S  (A=EMw, B^T=S^T)
  gemm_bf16_ABt<<<gemmNN, blk256, 0, stream>>>(bf1, bf2, G, nullptr,
                                               Nn, Nn, Nn, Nn, Nn, Nn);
  // 11) U^T (bf16), reusing bf3
  k_transU<<<tgrid, tblk, 0, stream>>>(G, bf3);
  // 12) new_w = S^T @ U  (A=S^T, B^T=U^T); epilogue also emits new_adj
  gemm_bf16_ABt<<<gemmNN, blk256, 0, stream>>>(bf2, bf3, out_neww, out_adj,
                                               Nn, Nn, Nn, Nn, Nn, Nn);
}
